// ClusterHead_66589172957621
// MI455X (gfx1250) — matrix operand order repaired and checked by reading, not compiled, not run on hardware
//
#include <hip/hip_runtime.h>
#include <hip/hip_bf16.h>

// CDNA5 / gfx1250, wave32. D = A*B + C via v_wmma_f32_16x16x32_bf16.
typedef __attribute__((ext_vector_type(16))) __bf16 v16bf;
typedef __attribute__((ext_vector_type(8)))  float  v8f;

constexpr int N = 32768;   // rows of x
constexpr int D = 512;     // reduction dim
constexpr int K = 1024;    // clusters
constexpr int KSTEPS = D / 32;   // 16 wmma k-steps
constexpr int CTILES = K / 16;   // 64 column tiles
constexpr int ROWS_PER_WG = 32;

// ---------------------------------------------------------------------------
// Prep kernel 1: pack centers (fp32 [K][D] row-major) into bf16 WMMA-B
// fragments, laid out [ctile(64)][kstep(16)][lane(32)][elem(16)] so the GEMM
// hot loop is pure contiguous 32B-per-lane loads (2x global_load_b128).
// B k order chosen here (contiguous, not AMD's table order; A uses the same): lane L holds column N = L&15,
// depth k = elem + (L>=16 ? 16 : 0), contiguous 16 depth values.
// ---------------------------------------------------------------------------
__global__ __launch_bounds__(256) void pack_centers(const float* __restrict__ c,
                                                    __bf16* __restrict__ cbf) {
  int idx  = blockIdx.x * 256 + threadIdx.x;   // 0 .. 64*16*32-1 = 32767
  int lane = idx & 31;
  int ks   = (idx >> 5) & (KSTEPS - 1);
  int ct   = idx >> 9;
  const float* src = c + (size_t)(ct * 16 + (lane & 15)) * D
                       + ks * 32 + ((lane >> 4) << 4);
  v16bf v;
  #pragma unroll
  for (int i = 0; i < 16; ++i) v[i] = (__bf16)src[i];
  *(v16bf*)(cbf + ((size_t)idx << 4)) = v;
}

// Prep kernel 2: hcsq[k] = 0.5 * ||c_k||^2  (the softmax-invariant bias).
__global__ __launch_bounds__(256) void center_norms(const float* __restrict__ c,
                                                    float* __restrict__ hcsq) {
  int k = blockIdx.x * 256 + threadIdx.x;      // 0..1023
  const float4* row = (const float4*)(c + (size_t)k * D);
  float s = 0.f;
  #pragma unroll 4
  for (int i = 0; i < D / 4; ++i) {
    float4 v = row[i];
    s += v.x * v.x + v.y * v.y + v.z * v.z + v.w * v.w;
  }
  hcsq[k] = 0.5f * s;
}

// ---------------------------------------------------------------------------
// Row softmax: one wave per 1024-wide row. lane reads elems lane+32k (bank
// conflict free), reduces max/sum with wave32 shfl_xor, writes coalesced.
// ---------------------------------------------------------------------------
__device__ __forceinline__ void softmax_row(const float* __restrict__ lrow,
                                            float* __restrict__ orow, int lane) {
  float v[32];
  float mx = -3.402823466e38f;
  #pragma unroll
  for (int k = 0; k < 32; ++k) { v[k] = lrow[lane + 32 * k]; mx = fmaxf(mx, v[k]); }
  #pragma unroll
  for (int off = 16; off >= 1; off >>= 1) mx = fmaxf(mx, __shfl_xor(mx, off, 32));
  float s = 0.f;
  #pragma unroll
  for (int k = 0; k < 32; ++k) { v[k] = __expf(v[k] - mx); s += v[k]; }
  #pragma unroll
  for (int off = 16; off >= 1; off >>= 1) s += __shfl_xor(s, off, 32);
  float inv = 1.0f / s;
  #pragma unroll
  for (int k = 0; k < 32; ++k) orow[lane + 32 * k] = v[k] * inv;
}

__device__ __forceinline__ void store_rows(float* __restrict__ lds,
                                           const v8f* __restrict__ acc8,
                                           const float* __restrict__ bias,
                                           int w, int m) {
  #pragma unroll
  for (int j = 0; j < 8; ++j) {
    int col = w * 128 + j * 16 + m;   // 16 consecutive dwords per half-wave: conflict free
    #pragma unroll
    for (int r = 0; r < 8; ++r) lds[r * K + col] = acc8[j][r] - bias[j];
  }
}

// ---------------------------------------------------------------------------
// Main kernel: 256 threads (8 waves). WG computes rows [32b, 32b+32) x all K.
// Wave w owns columns [w*128, w*128+128): 8 C tiles x 2 row fragments.
// ---------------------------------------------------------------------------
__global__ __launch_bounds__(256) void cluster_head(const float* __restrict__ x,
                                                    const __bf16* __restrict__ cbf,
                                                    const float* __restrict__ hcsq,
                                                    float* __restrict__ out) {
  __shared__ float lds[8 * K];                 // 32 KB staging for 8 rows

  const int lane    = threadIdx.x & 31;
  const int w       = threadIdx.x >> 5;
  const int m       = lane & 15;
  const int hi      = lane >> 4;               // 0: rows 0-7, 1: rows 8-15 of a tile
  const int rowbase = blockIdx.x * ROWS_PER_WG;

  const float* xrow0 = x + (size_t)(rowbase + m) * D;
  const float* xrow1 = x + (size_t)(rowbase + 16 + m) * D;

  v8f accA[8] = {};
  v8f accB[8] = {};

  for (int ks = 0; ks < KSTEPS; ++ks) {
    // A fragment, k order chosen here (contiguous, not AMD's table order): lane<16 -> K in {d0..d0+15}
    // with d0 = ks*32; lanes>=16 shifted by 16 (matches packed B).
    const int d0 = ks * 32 + hi * 16;
    float4 p0 = *(const float4*)(xrow0 + d0);
    float4 p1 = *(const float4*)(xrow0 + d0 + 4);
    float4 p2 = *(const float4*)(xrow0 + d0 + 8);
    float4 p3 = *(const float4*)(xrow0 + d0 + 12);
    float4 q0 = *(const float4*)(xrow1 + d0);
    float4 q1 = *(const float4*)(xrow1 + d0 + 4);
    float4 q2 = *(const float4*)(xrow1 + d0 + 8);
    float4 q3 = *(const float4*)(xrow1 + d0 + 12);
    v16bf aA, aB;
    aA[0]=(__bf16)p0.x; aA[1]=(__bf16)p0.y; aA[2]=(__bf16)p0.z; aA[3]=(__bf16)p0.w;
    aA[4]=(__bf16)p1.x; aA[5]=(__bf16)p1.y; aA[6]=(__bf16)p1.z; aA[7]=(__bf16)p1.w;
    aA[8]=(__bf16)p2.x; aA[9]=(__bf16)p2.y; aA[10]=(__bf16)p2.z; aA[11]=(__bf16)p2.w;
    aA[12]=(__bf16)p3.x; aA[13]=(__bf16)p3.y; aA[14]=(__bf16)p3.z; aA[15]=(__bf16)p3.w;
    aB[0]=(__bf16)q0.x; aB[1]=(__bf16)q0.y; aB[2]=(__bf16)q0.z; aB[3]=(__bf16)q0.w;
    aB[4]=(__bf16)q1.x; aB[5]=(__bf16)q1.y; aB[6]=(__bf16)q1.z; aB[7]=(__bf16)q1.w;
    aB[8]=(__bf16)q2.x; aB[9]=(__bf16)q2.y; aB[10]=(__bf16)q2.z; aB[11]=(__bf16)q2.w;
    aB[12]=(__bf16)q3.x; aB[13]=(__bf16)q3.y; aB[14]=(__bf16)q3.z; aB[15]=(__bf16)q3.w;

    #pragma unroll
    for (int j = 0; j < 8; ++j) {
      const int ct = w * 8 + j;
      v16bf b = *(const v16bf*)(cbf + ((size_t)((ct * KSTEPS + ks) * 32 + lane) << 4));
      accA[j] = __builtin_amdgcn_wmma_f32_16x16x32_bf16(
          false, aA, false, b, (short)0, accA[j], false, false);
      accB[j] = __builtin_amdgcn_wmma_f32_16x16x32_bf16(
          false, aB, false, b, (short)0, accB[j], false, false);
    }
  }

  float bias[8];
  #pragma unroll
  for (int j = 0; j < 8; ++j) bias[j] = hcsq[w * 128 + j * 16 + m];

  // Four phases of 8 rows each: stage logits in LDS, softmax, write out.
  if (hi == 0) store_rows(lds, accA, bias, w, m);
  __syncthreads();
  softmax_row(lds + w * K, out + (size_t)(rowbase + w) * K, lane);
  __syncthreads();

  if (hi == 1) store_rows(lds, accA, bias, w, m);
  __syncthreads();
  softmax_row(lds + w * K, out + (size_t)(rowbase + 8 + w) * K, lane);
  __syncthreads();

  if (hi == 0) store_rows(lds, accB, bias, w, m);
  __syncthreads();
  softmax_row(lds + w * K, out + (size_t)(rowbase + 16 + w) * K, lane);
  __syncthreads();

  if (hi == 1) store_rows(lds, accB, bias, w, m);
  __syncthreads();
  softmax_row(lds + w * K, out + (size_t)(rowbase + 24 + w) * K, lane);
}

extern "C" void kernel_launch(void* const* d_in, const int* in_sizes, int n_in,
                              void* d_out, int out_size, void* d_ws, size_t ws_size,
                              hipStream_t stream) {
  const float* x       = (const float*)d_in[0];   // [N, D] fp32
  const float* centers = (const float*)d_in[1];   // [K, D] fp32
  float* out           = (float*)d_out;           // [N, K] fp32

  // Workspace: 1 MB packed bf16 centers + 4 KB half-norms.
  __bf16* cbf = (__bf16*)d_ws;
  float* hcsq = (float*)((char*)d_ws + (size_t)K * D * sizeof(__bf16));

  pack_centers<<<(CTILES * KSTEPS * 32) / 256, 256, 0, stream>>>(centers, cbf);
  center_norms<<<K / 256, 256, 0, stream>>>(centers, hcsq);
  cluster_head<<<N / ROWS_PER_WG, 256, 0, stream>>>(x, cbf, hcsq, out);
}
